// GATLayer_56221121904663
// MI455X (gfx1250) — matrix operand order repaired and checked by reading, not compiled, not run on hardware
//
#include <hip/hip_runtime.h>
#include <hip/hip_bf16.h>

// ---------------------------------------------------------------------------
// GAT layer for MI455X (gfx1250): B=8, S=8192, E=32768, IN_F=OUT_F=256, ED=64
// ---------------------------------------------------------------------------
#define GB   8
#define GS   8192
#define GE   32768
#define GIN  256
#define GOUT 256
#define GED  64

typedef __attribute__((ext_vector_type(16))) __bf16 v16bf;
typedef __attribute__((ext_vector_type(8)))  float  v8f;

// ---------------------------------------------------------------------------
// Kernel 1: h = (x @ W) with node_mask zeroing, via v_wmma_f32_16x16x32_bf16.
//
// block = 256 threads = 8 waves. W staged once per block into LDS as bf16,
// transposed to Wt[n][k] (128 KB of 320 KB WGP LDS): a B fragment (col n,
// 16 consecutive K) is one contiguous 32-byte LDS read (ds_load_b128 x2,
// conflict-free). Each wave computes a 16(M) x 256(N) strip: one A fragment
// feeds 16 WMMAs; x rows are read from HBM exactly once.
//
// Scheduling: B fragments software double-buffered (load t+1 before WMMA t);
// next k-step's A fetched before the 16-WMMA burst. The second LDS base
// pointer (64 KB up) is laundered through empty inline asm so the optimizer
// cannot re-fold it onto the first base: all 16 tiles then use in-range
// 16-bit DS immediates and no VALU address math lands in the WMMA hazard
// window (no v_or / v_nop runs).
// ---------------------------------------------------------------------------
__global__ void __launch_bounds__(256)
gat_gemm_wmma(const float* __restrict__ x, const float* __restrict__ W,
              const unsigned char* __restrict__ node_mask,
              float* __restrict__ h)
{
    __shared__ alignas(32) __bf16 Wt[GOUT][GIN];   // [n][k], 128 KB

    const int tid  = threadIdx.x;
    const int lane = tid & 31;
    const int wave = tid >> 5;
    const int half = lane >> 4;                    // 0 or 1
    const int lrow = lane & 15;

    // ---- stage W -> LDS (bf16, transposed). Coalesced float4 global reads.
    for (int base = tid * 4; base < GIN * GOUT; base += 256 * 4) {
        const int k = base >> 8;                   // base = k*256 + n
        const int n = base & 255;
        const float4 w = *reinterpret_cast<const float4*>(W + base);
        Wt[n + 0][k] = (__bf16)w.x;
        Wt[n + 1][k] = (__bf16)w.y;
        Wt[n + 2][k] = (__bf16)w.z;
        Wt[n + 3][k] = (__bf16)w.w;
    }
    __syncthreads();

    // ---- wave-level 16x256 strip
    const long rowA = (long)blockIdx.x * 128 + wave * 16 + lrow;  // A: M=lane%16
    const float* __restrict__ xr = x + rowA * GIN;

    v8f acc[16];
    #pragma unroll
    for (int t = 0; t < 16; ++t) acc[t] = (v8f){};

    // opaque 64 KB byte offset: keeps the second DS base in its own register
    unsigned hi_bytes = 8u * 16u * GIN * sizeof(__bf16);   // 0x10000
    asm("" : "+v"(hi_bytes));

    // preload A data for k0 = 0
    float av[16];
    {
        const int ka0 = half * 16;
        const int ka1 = half * 16 + 8;
        *reinterpret_cast<float4*>(&av[0])  = *reinterpret_cast<const float4*>(xr + ka0);
        *reinterpret_cast<float4*>(&av[4])  = *reinterpret_cast<const float4*>(xr + ka0 + 4);
        *reinterpret_cast<float4*>(&av[8])  = *reinterpret_cast<const float4*>(xr + ka1);
        *reinterpret_cast<float4*>(&av[12]) = *reinterpret_cast<const float4*>(xr + ka1 + 4);
    }

    #pragma unroll
    for (int k0 = 0; k0 < GIN; k0 += 32) {
        // ---- pack current A fragment (contiguous order: element j is k = k0 + 16*half + j, same as B)
        v16bf afrag;
        #pragma unroll
        for (int j = 0; j < 16; ++j) afrag[j] = (__bf16)av[j];

        // ---- issue next k-step's A loads now: ~16 WMMAs of latency cover
        if (k0 + 32 < GIN) {
            const int ka0 = (k0 + 32) + half * 16;
            const int ka1 = (k0 + 32) + half * 16 + 8;
            *reinterpret_cast<float4*>(&av[0])  = *reinterpret_cast<const float4*>(xr + ka0);
            *reinterpret_cast<float4*>(&av[4])  = *reinterpret_cast<const float4*>(xr + ka0 + 4);
            *reinterpret_cast<float4*>(&av[8])  = *reinterpret_cast<const float4*>(xr + ka1);
            *reinterpret_cast<float4*>(&av[12]) = *reinterpret_cast<const float4*>(xr + ka1 + 4);
        }

        // ---- 16 N-tiles share this A fragment; B double-buffered from LDS.
        // wb0 serves tiles 0..7, wb1 (opaque +64 KB) serves tiles 8..15, so
        // every ds_load offset fits the 16-bit DS immediate.
        const int kb0 = k0 + half * 16;            // B: K = kb0 .. kb0+15
        const __bf16* __restrict__ wb0 = &Wt[lrow][kb0];
        const __bf16* __restrict__ wb1 =
            reinterpret_cast<const __bf16*>(
                reinterpret_cast<const char*>(wb0) + hi_bytes);
        v16bf bcur = *reinterpret_cast<const v16bf*>(wb0);
        #pragma unroll
        for (int t = 0; t < 16; ++t) {
            v16bf bnxt = bcur;
            if (t < 15) {
                const int u = t + 1;               // compile-time constant
                const __bf16* p = (u < 8) ? (wb0 + u * 16 * GIN)
                                          : (wb1 + (u - 8) * 16 * GIN);
                bnxt = *reinterpret_cast<const v16bf*>(p);
            }
            acc[t] = __builtin_amdgcn_wmma_f32_16x16x32_bf16(
                         false, afrag, false, bcur, (short)0, acc[t],
                         false, false);
            bcur = bnxt;
        }
    }

    // ---- C/D store: lanes 0-15 -> M=r, lanes 16-31 -> M=r+8 ; N = lane%16
    const long rowbase = (long)blockIdx.x * 128 + wave * 16;
    #pragma unroll
    for (int r = 0; r < 8; ++r) {
        const long row = rowbase + r + half * 8;
        const bool msk = node_mask[row] != 0;
        float* __restrict__ hr = h + row * GOUT + lrow;
        #pragma unroll
        for (int t = 0; t < 16; ++t)
            hr[t * 16] = msk ? 0.0f : acc[t][r];
    }
}

// ---------------------------------------------------------------------------
// Kernel 2: per-edge attention logits, wave32 per edge.
// e = leaky_relu(h_src.a_src + h_dst.a_dst + edge_attr.a_edge), mask -> -inf
// ---------------------------------------------------------------------------
__global__ void __launch_bounds__(256)
gat_edge_score(const float* __restrict__ h, const int* __restrict__ edge_index,
               const float* __restrict__ edge_attr,
               const unsigned char* __restrict__ edge_mask,
               const float* __restrict__ a, float* __restrict__ e)
{
    const int wid  = (blockIdx.x * blockDim.x + threadIdx.x) >> 5;
    const int lane = threadIdx.x & 31;
    if (wid >= GB * GE) return;
    const int b = wid / GE;

    const int src = edge_index[(size_t)wid * 2 + 0];
    const int dst = edge_index[(size_t)wid * 2 + 1];
    const float* __restrict__ hs = h + ((size_t)b * GS + src) * GOUT;
    const float* __restrict__ hd = h + ((size_t)b * GS + dst) * GOUT;
    const float* __restrict__ ea = edge_attr + (size_t)wid * GED;

    // pull the gathered rows toward the WGP (h is L2-resident: 64MB < 192MB L2)
    __builtin_prefetch(hs + lane * 4, 0, 0);
    __builtin_prefetch(hd + lane * 4, 0, 0);

    float acc = 0.0f;
    #pragma unroll
    for (int f = lane; f < GOUT; f += 32)
        acc = fmaf(hs[f], a[f], fmaf(hd[f], a[GOUT + f], acc));
    #pragma unroll
    for (int f = lane; f < GED; f += 32)
        acc = fmaf(ea[f], a[2 * GOUT + f], acc);

    #pragma unroll
    for (int off = 16; off > 0; off >>= 1)
        acc += __shfl_xor(acc, off, 32);

    if (lane == 0) {
        float v = (acc >= 0.0f) ? acc : 0.2f * acc;   // leaky_relu slope 0.2
        if (edge_mask[wid]) v = -__builtin_inff();
        e[wid] = v;
    }
}

// ---------------------------------------------------------------------------
// Kernel 3: masked softmax over E per batch. One 1024-thread block per batch.
// ---------------------------------------------------------------------------
__global__ void __launch_bounds__(1024)
gat_softmax(const float* __restrict__ e, const unsigned char* __restrict__ edge_mask,
            float* __restrict__ alpha)
{
    __shared__ float sred[32];
    const int b    = blockIdx.x;
    const int lane = threadIdx.x & 31;
    const int wv   = threadIdx.x >> 5;
    const float* __restrict__ eb = e + (size_t)b * GE;

    // --- pass 1: max ---
    float m = -__builtin_inff();
    for (int i = threadIdx.x; i < GE; i += 1024) m = fmaxf(m, eb[i]);
    #pragma unroll
    for (int o = 16; o > 0; o >>= 1) m = fmaxf(m, __shfl_xor(m, o, 32));
    if (lane == 0) sred[wv] = m;
    __syncthreads();
    m = sred[lane];                                  // 32 waves -> 32 entries
    #pragma unroll
    for (int o = 16; o > 0; o >>= 1) m = fmaxf(m, __shfl_xor(m, o, 32));
    m = __shfl(m, 0, 32);
    if (m == -__builtin_inff()) m = 0.0f;            // fully-masked guard
    __syncthreads();

    // --- pass 2: sum of exp ---
    float s = 0.0f;
    for (int i = threadIdx.x; i < GE; i += 1024) s += __expf(eb[i] - m);
    #pragma unroll
    for (int o = 16; o > 0; o >>= 1) s += __shfl_xor(s, o, 32);
    if (lane == 0) sred[wv] = s;
    __syncthreads();
    s = sred[lane];
    #pragma unroll
    for (int o = 16; o > 0; o >>= 1) s += __shfl_xor(s, o, 32);
    s = __shfl(s, 0, 32);
    const float inv = 1.0f / s;

    // --- pass 3: write alpha ---
    for (int i = threadIdx.x; i < GE; i += 1024) {
        float av = __expf(eb[i] - m) * inv;
        if (edge_mask[(size_t)b * GE + i]) av = 0.0f;
        alpha[(size_t)b * GE + i] = av;
    }
}

// ---------------------------------------------------------------------------
// Kernel 4: zero output (float4 stores)
// ---------------------------------------------------------------------------
__global__ void __launch_bounds__(256)
gat_zero(float4* __restrict__ out, long n4)
{
    long i = (long)blockIdx.x * blockDim.x + threadIdx.x;
    if (i < n4) out[i] = make_float4(0.f, 0.f, 0.f, 0.f);
}

// ---------------------------------------------------------------------------
// Kernel 5: scatter msg = alpha * h_src into out[dst] via f32 atomics
// (out is 64MB -> L2-resident on the 192MB L2; atomics resolve in L2)
// ---------------------------------------------------------------------------
__global__ void __launch_bounds__(256)
gat_scatter(const float* __restrict__ h, const int* __restrict__ edge_index,
            const float* __restrict__ alpha, float* __restrict__ out)
{
    const int wid  = (blockIdx.x * blockDim.x + threadIdx.x) >> 5;
    const int lane = threadIdx.x & 31;
    if (wid >= GB * GE) return;
    const int b   = wid / GE;
    const int src = edge_index[(size_t)wid * 2 + 0];
    const int dst = edge_index[(size_t)wid * 2 + 1];
    const float al = alpha[wid];

    const float* __restrict__ hs = h + ((size_t)b * GS + src) * GOUT;
    float* __restrict__ ob = out + ((size_t)b * GS + dst) * GOUT;
    __builtin_prefetch(hs + lane * 4, 0, 0);

    #pragma unroll
    for (int f = lane; f < GOUT; f += 32)
        atomicAdd(&ob[f], al * hs[f]);
}

// ---------------------------------------------------------------------------
// Kernel 6: apply node_mask to output
// ---------------------------------------------------------------------------
__global__ void __launch_bounds__(256)
gat_node_mask(float* __restrict__ out, const unsigned char* __restrict__ node_mask)
{
    long i = (long)blockIdx.x * blockDim.x + threadIdx.x;   // over B*S*GOUT
    if (i < (long)GB * GS * GOUT && node_mask[i / GOUT]) out[i] = 0.0f;
}

// ---------------------------------------------------------------------------
extern "C" void kernel_launch(void* const* d_in, const int* in_sizes, int n_in,
                              void* d_out, int out_size, void* d_ws, size_t ws_size,
                              hipStream_t stream)
{
    const float*         x          = (const float*)d_in[0];         // B*S*IN_F
    const int*           edge_index = (const int*)d_in[1];           // B*E*2
    const float*         edge_attr  = (const float*)d_in[2];         // B*E*ED
    const unsigned char* node_mask  = (const unsigned char*)d_in[3]; // B*S bool
    const unsigned char* edge_mask  = (const unsigned char*)d_in[4]; // B*E bool
    const float*         W          = (const float*)d_in[5];         // IN_F*OUT_F
    const float*         a          = (const float*)d_in[6];         // 2*OUT_F+ED
    float* out = (float*)d_out;                                      // B*S*OUT_F

    // workspace layout
    float* h     = (float*)d_ws;                       // B*S*OUT_F  (64 MB)
    float* e     = h + (size_t)GB * GS * GOUT;         // B*E        (1 MB)
    float* alpha = e + (size_t)GB * GE;                // B*E        (1 MB)

    // K1: WMMA GEMM. block = 8 waves, each wave a 16x256 strip -> 128 rows/blk.
    {
        const int blocks = (GB * GS) / 128;                // 512
        gat_gemm_wmma<<<blocks, 256, 0, stream>>>(x, W, node_mask, h);
    }
    // K2: edge scores. one wave per edge.
    {
        const int waves  = GB * GE;                        // 262144
        const int blocks = (waves * 32 + 255) / 256;       // 32768
        gat_edge_score<<<blocks, 256, 0, stream>>>(h, edge_index, edge_attr,
                                                   edge_mask, a, e);
    }
    // K3: softmax per batch.
    gat_softmax<<<GB, 1024, 0, stream>>>(e, edge_mask, alpha);

    // K4: zero out.
    {
        const long n4 = (long)GB * GS * GOUT / 4;          // 4M float4
        gat_zero<<<(int)((n4 + 255) / 256), 256, 0, stream>>>((float4*)out, n4);
    }
    // K5: scatter-add.
    {
        const int waves  = GB * GE;
        const int blocks = (waves * 32 + 255) / 256;
        gat_scatter<<<blocks, 256, 0, stream>>>(h, edge_index, alpha, out);
    }
    // K6: node mask on output.
    {
        const long n = (long)GB * GS * GOUT;
        gat_node_mask<<<(int)((n + 255) / 256), 256, 0, stream>>>(out, node_mask);
    }
}
